// MultiHeadAttention_63153199120543
// MI455X (gfx1250) — compile-verified, matrix operand order checked by reading, not run on hardware
//
#include <hip/hip_runtime.h>

// ---------------------------------------------------------------------------
// MultiHeadAttention block for MI455X (gfx1250, wave32, WMMA + TDM).
//   S=8, B=1024, E=768, H=12, DK=DV=64.  M = S*B = 8192.
// GEMMs on v_wmma_f32_16x16x32_f16; shared A-stripes staged to LDS once per
// block with tensor_load_to_lds (TENSORcnt) instead of 8x redundant global
// loads.
// ---------------------------------------------------------------------------

typedef __attribute__((ext_vector_type(16))) _Float16     v16h;
typedef __attribute__((ext_vector_type(8)))  _Float16     v8h;
typedef __attribute__((ext_vector_type(8)))  float        v8f;
typedef __attribute__((ext_vector_type(4)))  unsigned int uint4v;
typedef __attribute__((ext_vector_type(8)))  int          int8v;
typedef __attribute__((ext_vector_type(4)))  int          int4v;

static constexpr int S_ = 8;
static constexpr int B_ = 1024;
static constexpr int E_ = 768;
static constexpr int H_ = 12;
static constexpr int D_ = 64;
static constexpr int M_ = S_ * B_;        // 8192 rows (s,b) flattened

// ---------------------------------------------------------------------------
// Fragment loaders per ISA 7.12.2 (16-bit operand 16x32 layout, wave32):
//   lane L: row/col = L&15, half-select hs = L>>4.
//   elements 0..7  -> k = kk + hs*8 + i ; elements 8..15 -> k = kk+16+hs*8+i
// Works for both global and LDS (generic) pointers -> b128 loads.
// ---------------------------------------------------------------------------
__device__ __forceinline__ v16h load_frag_h(const _Float16* rowk, int hs) {
    union { v16h v; v8h h[2]; } u;
    u.h[0] = *(const v8h*)(rowk + hs * 8);
    u.h[1] = *(const v8h*)(rowk + 16 + hs * 8);
    return u.v;
}

// Fragment with f32 -> f16 conversion (attention probabilities).
__device__ __forceinline__ v16h load_frag_f32(const float* rowk, int hs) {
    union { v16h v; _Float16 h[16]; } u;
    const float* p0 = rowk + hs * 8;
    const float* p1 = rowk + 16 + hs * 8;
#pragma unroll
    for (int i = 0; i < 8; ++i) u.h[i] = (_Float16)p0[i];
#pragma unroll
    for (int i = 0; i < 8; ++i) u.h[8 + i] = (_Float16)p1[i];
    return u.v;
}

__device__ __forceinline__ v8f wmma16(v16h a, v16h b, v8f c) {
    // (neg_a, A, neg_b, B, c_mod, C, reuse_a, reuse_b)
    return __builtin_amdgcn_wmma_f32_16x16x32_f16(false, a, false, b, (short)0, c,
                                                  false, false);
}

__device__ __forceinline__ v8f vzero8() {
    v8f z = {0.f, 0.f, 0.f, 0.f, 0.f, 0.f, 0.f, 0.f};
    return z;
}

// ---------------------------------------------------------------------------
// Tensor Data Mover: DMA a 2D tile of f16 (data_size=2B) from global to LDS.
// Descriptor packing per CDNA5 ISA ch.8 (D# group0 128b + group1 256b;
// groups 2/3 zero for a <=2D tensor).  Issued by one wave per block.
// 6-arg builtin form (clang-23 / therock-10.0 headers).
// ---------------------------------------------------------------------------
__device__ __forceinline__ void tdm_load_tile_2d(const void* gsrc,
                                                 unsigned lds_off,
                                                 unsigned tensor_d0,
                                                 unsigned tensor_d1,
                                                 unsigned tile_d0,
                                                 unsigned tile_d1,
                                                 unsigned long long d0_stride)
{
    unsigned long long ga = (unsigned long long)gsrc;
    uint4v g0;
    g0.x = 1u;                                                // count=1, user D#
    g0.y = lds_off;                                           // lds_addr[31:0]
    g0.z = (unsigned)ga;                                      // global_addr[31:0]
    g0.w = (unsigned)((ga >> 32) & 0x01FFFFFFu) | (2u << 30); // ga[56:32] | type=2
    int8v g1;
    g1[0] = (int)(1u << 16);                                  // mask=0, data_size=1 (2B)
    g1[1] = (int)((tensor_d0 & 0xFFFFu) << 16);               // tensor_dim0[15:0] @63:48
    g1[2] = (int)((tensor_d0 >> 16) | ((tensor_d1 & 0xFFFFu) << 16)); // d0 hi | d1 lo
    g1[3] = (int)((tensor_d1 >> 16) | (tile_d0 << 16));       // d1 hi | tile_dim0
    g1[4] = (int)tile_d1;                                     // tile_dim1 (tile_dim2=0)
    g1[5] = (int)(unsigned)(d0_stride & 0xFFFFFFFFu);         // dim0_stride[31:0]
    g1[6] = (int)(unsigned)((d0_stride >> 32) & 0xFFFFu);     // dim0_stride[47:32]
    g1[7] = 0;
    int4v gz4 = {0, 0, 0, 0};
    int8v gz8 = {0, 0, 0, 0, 0, 0, 0, 0};
    __builtin_amdgcn_tensor_load_to_lds(g0, g1, gz4, gz4, gz8, 0);
}

__device__ __forceinline__ unsigned lds_offset_of(const void* p) {
    // LDS aperture: flat addr[31:0] is the LDS offset (ISA 10.2).
    return (unsigned)(unsigned long long)p;
}

// ---------------------------------------------------------------------------
// Kernel 0: f32 -> f16 conversion (grid-stride)
// ---------------------------------------------------------------------------
__global__ void __launch_bounds__(256)
cvt_f32_to_f16_kernel(const float* __restrict__ src, _Float16* __restrict__ dst, int n) {
    int i = blockIdx.x * blockDim.x + threadIdx.x;
    int stride = gridDim.x * blockDim.x;
    for (; i < n; i += stride) dst[i] = (_Float16)src[i];
}

// ---------------------------------------------------------------------------
// Kernel 1: projection GEMM  out = X @ W^T + bias
//   Block = one 16-row stripe x 8 consecutive N tiles (8 waves).
//   A stripe (16 x 768 f16, shared by all 8 waves) staged via TDM into LDS.
//   vmode==0: store q/k layout [s,h,b,d]; vmode==1: store v transposed [s,h,d,b].
// ---------------------------------------------------------------------------
__global__ void __launch_bounds__(256)
proj_gemm_kernel(const _Float16* __restrict__ X,
                 const _Float16* __restrict__ W,
                 const float*    __restrict__ bias,
                 _Float16*       __restrict__ out,
                 int vmode)
{
    __shared__ _Float16 astage[16 * E_];          // 24 KB

    const int lane = threadIdx.x & 31;
    const int wave = threadIdx.x >> 5;
    const int tm   = blockIdx.x / 6;              // 512 row stripes
    const int tn   = (blockIdx.x % 6) * 8 + wave; // 48 N tiles
    const int hs   = lane >> 4;
    const int l16  = lane & 15;

    if (wave == 0) {
        tdm_load_tile_2d(X + (size_t)tm * 16 * E_, lds_offset_of(astage),
                         E_, M_, E_, 16, E_);
        __builtin_amdgcn_s_wait_tensorcnt(0);
    }
    __syncthreads();

    const _Float16* Arow = astage + (size_t)l16 * E_;          // LDS
    const _Float16* Brow = W + (size_t)(tn * 16 + l16) * E_;   // global (L2-resident)

    v8f acc = vzero8();
#pragma unroll 4
    for (int kk = 0; kk < E_; kk += 32) {
        __builtin_prefetch(Brow + kk + 128, 0, 0);   // global_prefetch_b8
        v16h a = load_frag_h(Arow + kk, hs);
        v16h b = load_frag_h(Brow + kk, hs);
        acc = wmma16(a, b, acc);
    }

    const int   n  = tn * 16 + l16;
    const float bn = bias[n];
    const int   h  = n >> 6;          // head
    const int   d  = n & 63;          // dim within head
#pragma unroll
    for (int r = 0; r < 8; ++r) {
        int m = tm * 16 + r + 8 * hs;              // C layout: row = r + 8*hs
        int s = m >> 10, bb = m & 1023;
        size_t addr = vmode
            ? ((size_t)((s * H_ + h) * D_ + d) * B_ + bb)    // [s,h,d,b]
            : ((size_t)((s * H_ + h) * B_ + bb) * D_ + d);   // [s,h,b,d]
        out[addr] = (_Float16)(acc[r] + bn);
    }
}

// ---------------------------------------------------------------------------
// Kernel 2: scores + softmax, fused.
//   qp, kp: [S*H, B, D] f16.   attn out: [S*H, B, B] f32.
//   Block = one (s,h, 16-row stripe); q stripe staged via TDM (shared by all
//   waves).  8 waves x 8 N-tiles = full 1024 cols; scores stay in registers;
//   softmax via __shfl_xor in 16-lane groups + small LDS across waves.
// ---------------------------------------------------------------------------
__global__ void __launch_bounds__(256)
scores_softmax_kernel(const _Float16* __restrict__ qp,
                      const _Float16* __restrict__ kp,
                      float*          __restrict__ attn)
{
    __shared__ _Float16 qstage[16 * D_];          // 2 KB
    __shared__ float    redmax[16][8];
    __shared__ float    redsum[16][8];

    const int lane = threadIdx.x & 31;
    const int wave = threadIdx.x >> 5;
    const int hs   = lane >> 4;
    const int l16  = lane & 15;
    const int tm   = blockIdx.x & 63;     // 64 row tiles
    const int sh   = blockIdx.x >> 6;     // 96 (s,h) pairs

    const _Float16* qbase = qp + (size_t)sh * B_ * D_;
    const _Float16* kbase = kp + (size_t)sh * B_ * D_;

    if (wave == 0) {
        tdm_load_tile_2d(qbase + (size_t)tm * 16 * D_, lds_offset_of(qstage),
                         D_, S_ * H_ * B_, D_, 16, D_);
        __builtin_amdgcn_s_wait_tensorcnt(0);
    }
    __syncthreads();

    // A fragments (16 rows x K=64) from LDS, reused across all 8 N-tiles.
    const _Float16* Arow = qstage + (size_t)l16 * D_;
    v16h a0 = load_frag_h(Arow,      hs);
    v16h a1 = load_frag_h(Arow + 32, hs);

    v8f c[8];
#pragma unroll
    for (int j = 0; j < 8; ++j) {
        const int nt = wave * 8 + j;
        const _Float16* Brow = kbase + (size_t)(nt * 16 + l16) * D_;
        v16h b0 = load_frag_h(Brow,      hs);
        v16h b1 = load_frag_h(Brow + 32, hs);
        v8f acc = vzero8();
        acc = wmma16(a0, b0, acc);
        acc = wmma16(a1, b1, acc);
#pragma unroll
        for (int r = 0; r < 8; ++r) acc[r] *= 0.125f;   // 1/sqrt(DK)
        c[j] = acc;
    }

    // -------- row max (this wave's 128 columns) --------
    float lmax[8];
#pragma unroll
    for (int r = 0; r < 8; ++r) {
        float v = c[0][r];
#pragma unroll
        for (int j = 1; j < 8; ++j) v = fmaxf(v, c[j][r]);
#pragma unroll
        for (int mask = 1; mask < 16; mask <<= 1)
            v = fmaxf(v, __shfl_xor(v, mask));
        lmax[r] = v;
    }
    if (l16 == 0) {
#pragma unroll
        for (int r = 0; r < 8; ++r) redmax[r + 8 * hs][wave] = lmax[r];
    }
    __syncthreads();
    float gmax[8];
#pragma unroll
    for (int r = 0; r < 8; ++r) {
        float v = redmax[r + 8 * hs][0];
#pragma unroll
        for (int w = 1; w < 8; ++w) v = fmaxf(v, redmax[r + 8 * hs][w]);
        gmax[r] = v;
    }

    // -------- exp + row sum --------
    float lsum[8];
#pragma unroll
    for (int r = 0; r < 8; ++r) lsum[r] = 0.f;
#pragma unroll
    for (int j = 0; j < 8; ++j) {
#pragma unroll
        for (int r = 0; r < 8; ++r) {
            float e = __expf(c[j][r] - gmax[r]);
            c[j][r] = e;
            lsum[r] += e;
        }
    }
#pragma unroll
    for (int r = 0; r < 8; ++r) {
        float v = lsum[r];
#pragma unroll
        for (int mask = 1; mask < 16; mask <<= 1)
            v += __shfl_xor(v, mask);
        lsum[r] = v;
    }
    if (l16 == 0) {
#pragma unroll
        for (int r = 0; r < 8; ++r) redsum[r + 8 * hs][wave] = lsum[r];
    }
    __syncthreads();
    float inv[8];
#pragma unroll
    for (int r = 0; r < 8; ++r) {
        float v = 0.f;
#pragma unroll
        for (int w = 0; w < 8; ++w) v += redsum[r + 8 * hs][w];
        inv[r] = 1.f / v;
    }

    // -------- write normalized probabilities --------
    float* abase = attn + (size_t)sh * B_ * B_;
#pragma unroll
    for (int j = 0; j < 8; ++j) {
        const int col = (wave * 8 + j) * 16 + l16;
#pragma unroll
        for (int r = 0; r < 8; ++r) {
            const int row = tm * 16 + r + 8 * hs;
            abase[(size_t)row * B_ + col] = c[j][r] * inv[r];
        }
    }
}

// ---------------------------------------------------------------------------
// Kernel 3: context = attn @ v.   attn f32 [S*H,B,B], vt f16 [S*H,D,B].
//   ctx out f16 [S, B, H*D]  (= [M_, E_] for the output projection).
//   One 16x16 tile per wave; A converted f32->f16 in-register (attn is the
//   dominant HBM stream, read once).
// ---------------------------------------------------------------------------
__global__ void __launch_bounds__(256)
context_gemm_kernel(const float*    __restrict__ attn,
                    const _Float16* __restrict__ vt,
                    _Float16*       __restrict__ ctx)
{
    const int lane = threadIdx.x & 31;
    const int wave = threadIdx.x >> 5;
    const int tile = blockIdx.x * 8 + wave;   // 96*64*4 = 24576 tiles
    const int tn   = tile & 3;                // 4 N tiles (D=64)
    const int tm   = (tile >> 2) & 63;        // 64 M tiles
    const int sh   = tile >> 8;               // 96
    const int h    = sh % H_;
    const int s    = sh / H_;
    const int hs   = lane >> 4;
    const int l16  = lane & 15;

    const float*    Arow = attn + (size_t)sh * B_ * B_ + (size_t)(tm * 16 + l16) * B_;
    const _Float16* Brow = vt   + (size_t)sh * D_ * B_ + (size_t)(tn * 16 + l16) * B_;

    v8f acc = vzero8();
#pragma unroll 2
    for (int kk = 0; kk < B_; kk += 32) {
        v16h a = load_frag_f32(Arow + kk, hs);
        v16h b = load_frag_h(Brow + kk, hs);
        acc = wmma16(a, b, acc);
    }

    const int d = tn * 16 + l16;
#pragma unroll
    for (int r = 0; r < 8; ++r) {
        const int bb = tm * 16 + r + 8 * hs;
        ctx[(size_t)(s * B_ + bb) * E_ + h * D_ + d] = (_Float16)acc[r];
    }
}

// ---------------------------------------------------------------------------
// Kernel 4: out = ctx @ Wo^T + bo + Q residual, then LayerNorm.  Fused.
//   Block = 16-row stripe (512 blocks). A stripe staged via TDM into LDS,
//   8 waves x 6 N-tiles = full E_=768.  The A-staging buffer is union'ed
//   with the LayerNorm stripe buffer (phases separated by barriers).
// ---------------------------------------------------------------------------
__global__ void __launch_bounds__(256)
out_ln_kernel(const _Float16* __restrict__ ctx,
              const _Float16* __restrict__ Wo,    // [E_, E_] (n,k)
              const float*    __restrict__ bo,
              const float*    __restrict__ Qin,   // residual, [M_, E_] f32
              const float*    __restrict__ g,
              const float*    __restrict__ bln,
              float*          __restrict__ outp)  // [M_, E_] f32
{
    __shared__ union {
        _Float16 a[16 * E_];                      // 24 KB: A staging (phase 1)
        float    x[16 * E_];                      // 48 KB: LN stripe (phase 2)
    } shm;

    const int lane = threadIdx.x & 31;
    const int wave = threadIdx.x >> 5;
    const int hs   = lane >> 4;
    const int l16  = lane & 15;
    const int tm   = blockIdx.x;                  // 512 row stripes

    if (wave == 0) {
        tdm_load_tile_2d(ctx + (size_t)tm * 16 * E_, lds_offset_of(shm.a),
                         E_, M_, E_, 16, E_);
        __builtin_amdgcn_s_wait_tensorcnt(0);
    }
    __syncthreads();

    const _Float16* Arow = shm.a + (size_t)l16 * E_;    // LDS

    v8f acc[6];
#pragma unroll
    for (int j = 0; j < 6; ++j) acc[j] = vzero8();

    for (int kk = 0; kk < E_; kk += 32) {
        v16h a = load_frag_h(Arow + kk, hs);      // shared by all 6 N-tiles
#pragma unroll
        for (int j = 0; j < 6; ++j) {
            const int n0 = (wave * 6 + j) * 16;
            const _Float16* Brow = Wo + (size_t)(n0 + l16) * E_ + kk;
            acc[j] = wmma16(a, load_frag_h(Brow, hs), acc[j]);
        }
    }
    __syncthreads();   // phase change: A staging done -> reuse LDS for LN stripe

    // bias + residual into LDS stripe
#pragma unroll
    for (int j = 0; j < 6; ++j) {
        const int n  = (wave * 6 + j) * 16 + l16;
        const float bn = bo[n];
#pragma unroll
        for (int r = 0; r < 8; ++r) {
            const int rl = r + 8 * hs;
            const int m  = tm * 16 + rl;
            shm.x[rl * E_ + n] = acc[j][r] + bn + Qin[(size_t)m * E_ + n];
        }
    }
    __syncthreads();

    // LayerNorm: 16 threads per row (consecutive -> same wave half).
    const int row = threadIdx.x >> 4;
    const int l   = threadIdx.x & 15;
    float sum = 0.f, sumsq = 0.f;
#pragma unroll 8
    for (int i = 0; i < E_ / 16; ++i) {
        float x = shm.x[row * E_ + l + 16 * i];
        sum   += x;
        sumsq += x * x;
    }
#pragma unroll
    for (int mask = 1; mask < 16; mask <<= 1) {
        sum   += __shfl_xor(sum,   mask);
        sumsq += __shfl_xor(sumsq, mask);
    }
    const float mu   = sum * (1.f / E_);
    const float var  = sumsq * (1.f / E_) - mu * mu;
    const float rstd = rsqrtf(var + 1e-5f);
    const int   m    = tm * 16 + row;
#pragma unroll 8
    for (int i = 0; i < E_ / 16; ++i) {
        const int cc = l + 16 * i;
        const float x = shm.x[row * E_ + cc];
        outp[(size_t)m * E_ + cc] = (x - mu) * rstd * g[cc] + bln[cc];
    }
}

// ---------------------------------------------------------------------------
// Host launcher
// ---------------------------------------------------------------------------
extern "C" void kernel_launch(void* const* d_in, const int* in_sizes, int n_in,
                              void* d_out, int out_size, void* d_ws, size_t ws_size,
                              hipStream_t stream) {
    const float* Q    = (const float*)d_in[0];
    const float* K    = (const float*)d_in[1];
    const float* V    = (const float*)d_in[2];
    const float* Wq   = (const float*)d_in[3];
    const float* bq   = (const float*)d_in[4];
    const float* Wk   = (const float*)d_in[5];
    const float* bk   = (const float*)d_in[6];
    const float* Wv   = (const float*)d_in[7];
    const float* bv   = (const float*)d_in[8];
    const float* Wo   = (const float*)d_in[9];
    const float* bo   = (const float*)d_in[10];
    const float* ln_g = (const float*)d_in[11];
    const float* ln_b = (const float*)d_in[12];

    // Output: [normed (S*B*E) | attn (S*H*B*B)] f32
    float* normed = (float*)d_out;
    float* attn   = (float*)d_out + (size_t)S_ * B_ * E_;

    // Workspace layout (all f16)
    const size_t NQ = (size_t)M_ * E_;          // 6291456
    const size_t NW = (size_t)E_ * E_;          // 589824
    const size_t NP = (size_t)S_ * H_ * B_ * D_;
    _Float16* w   = (_Float16*)d_ws;
    _Float16* Qh  = w;
    _Float16* Kh  = Qh  + NQ;
    _Float16* Vh  = Kh  + NQ;
    _Float16* Wqh = Vh  + NQ;
    _Float16* Wkh = Wqh + NW;
    _Float16* Wvh = Wkh + NW;
    _Float16* Woh = Wvh + NW;
    _Float16* qp  = Woh + NW;                   // [S*H, B, D]
    _Float16* kp  = qp  + NP;                   // [S*H, B, D]
    _Float16* vt  = kp  + NP;                   // [S*H, D, B] (transposed)
    _Float16* ctx = vt  + NP;                   // [M_, E_]

    auto cvt = [&](const float* s, _Float16* d, size_t n) {
        int blocks = (int)((n + 1023) / 1024);
        cvt_f32_to_f16_kernel<<<blocks, 256, 0, stream>>>(s, d, (int)n);
    };
    cvt(Q,  Qh,  NQ);
    cvt(K,  Kh,  NQ);
    cvt(V,  Vh,  NQ);
    cvt(Wq, Wqh, NW);
    cvt(Wk, Wkh, NW);
    cvt(Wv, Wvh, NW);
    cvt(Wo, Woh, NW);

    // QKV projections: 512 stripes x 6 N-groups = 3072 blocks
    proj_gemm_kernel<<<3072, 256, 0, stream>>>(Qh, Wqh, bq, qp, 0);
    proj_gemm_kernel<<<3072, 256, 0, stream>>>(Kh, Wkh, bk, kp, 0);
    proj_gemm_kernel<<<3072, 256, 0, stream>>>(Vh, Wvh, bv, vt, 1);

    // Scores + softmax: S*H * (B/16) = 6144 blocks
    scores_softmax_kernel<<<6144, 256, 0, stream>>>(qp, kp, attn);

    // Context: S*H * 64 * 4 tiles / 8 = 3072 blocks
    context_gemm_kernel<<<3072, 256, 0, stream>>>(attn, vt, ctx);

    // Output projection + residual + LayerNorm: M_/16 = 512 blocks
    out_ln_kernel<<<512, 256, 0, stream>>>(ctx, Woh, bo, Q, ln_g, ln_b, normed);
}
